// PermutedGru_17918603559096
// MI455X (gfx1250) — compile-verified, matrix operand order checked by reading, not run on hardware
//
#include <hip/hip_runtime.h>
#include <math.h>

// ---------------- constants ----------------
#define Hh   512
#define Tt   512
#define Bb   128
#define NROW (Tt * Bb)          // 65536 rows of X
#define KT   (Hh / 32)          // 16 k-tiles per GEMM
#define NT   (Hh / 16)          // 32 n-tiles
#define SCAN_BLOCKS 64

// workspace byte offsets
#define OFF_M0    ((size_t)0)
#define OFF_M     ((size_t)1 << 20)
#define OFF_P     ((size_t)2 << 20)
#define OFF_LOWER ((size_t)3 << 20)
#define OFF_WSW   ((size_t)4 << 20)                       // 6*512*512 bf16 = 3 MB
#define OFF_XB    ((size_t)8 << 20)                       // 65536*512 bf16 = 64 MB
#define OFF_XR    (OFF_XB + (size_t)NROW * Hh * 2)
#define OFF_XZ    (OFF_XR + (size_t)NROW * Hh * 4)
#define OFF_XN    (OFF_XZ + (size_t)NROW * Hh * 4)
#define OFF_H     (OFF_XN + (size_t)NROW * Hh * 4)
#define OFF_HBF   (OFF_H   + (size_t)Bb * Hh * 4)
#define OFF_ARH   (OFF_HBF + (size_t)Bb * Hh * 2)
#define OFF_Z     (OFF_ARH + (size_t)Bb * Hh * 2)
#define OFF_BAR   (OFF_Z   + (size_t)Bb * Hh * 4)

typedef __attribute__((ext_vector_type(16))) __bf16          v16bf;
typedef __attribute__((ext_vector_type(16))) unsigned short  v16us;
typedef __attribute__((ext_vector_type(8)))  unsigned short  v8us;
typedef __attribute__((ext_vector_type(8)))  float           v8f;
typedef __attribute__((ext_vector_type(4)))  unsigned int    u32x4;
typedef __attribute__((ext_vector_type(8)))  int             i32x8;
typedef __attribute__((ext_vector_type(4)))  int             i32x4;

#if defined(__has_builtin)
#if __has_builtin(__builtin_amdgcn_tensor_load_to_lds) && \
    __has_builtin(__builtin_amdgcn_s_wait_tensorcnt)
#define HAVE_TDM 1
#endif
#endif

__device__ __forceinline__ unsigned short f2bf(float f) {
  unsigned u = __builtin_bit_cast(unsigned, f);
  u += 0x7FFFu + ((u >> 16) & 1u);          // RNE
  return (unsigned short)(u >> 16);
}

// A fragment (16x32 bf16) from a row-major [rows][512] bf16 buffer.
// ISA layout: lane<16: M=lane, K=0..7 & 16..23 ; lane>=16: M=lane-16, K=8..15 & 24..31
__device__ __forceinline__ v16bf load_a_frag(const unsigned short* base, int m0, int k0) {
  const int lane = threadIdx.x & 31;
  const int mr   = lane & 15;
  const int kh   = (lane & 16) ? 8 : 0;
  const unsigned short* p = base + (size_t)(m0 + mr) * Hh + k0 + kh;
  v8us lo = *(const v8us*)p;
  v8us hi = *(const v8us*)(p + 16);
  v16us r;
#pragma unroll
  for (int i = 0; i < 8; ++i) { r[i] = lo[i]; r[i + 8] = hi[i]; }
  return __builtin_bit_cast(v16bf, r);
}

// B fragment from the pre-swizzled weight buffer in global memory.
__device__ __forceinline__ v16bf load_b_frag(const unsigned short* wsw, int w, int kt, int nt) {
  const int lane = threadIdx.x & 31;
  const unsigned short* p = wsw + ((((size_t)w * KT + kt) * NT + nt) * 32 + lane) * 16;
  return __builtin_bit_cast(v16bf, *(const v16us*)p);
}

// B fragment from LDS (pre-staged by TDM): slot-major [2][16 kt][32 lanes x 16]
__device__ __forceinline__ v16bf load_b_lds(const unsigned short* ldsW, int slot, int kt) {
  const int lane = threadIdx.x & 31;
  const unsigned short* p = ldsW + ((slot * KT + kt) * 32 + lane) * 16;
  return __builtin_bit_cast(v16bf, *(const v16us*)p);
}

__device__ __forceinline__ v8f wmma_bf16(v16bf a, v16bf b, v8f c) {
  return __builtin_amdgcn_wmma_f32_16x16x32_bf16(false, a, false, b, (short)0, c, false, false);
}

#if defined(HAVE_TDM)
// TDM: load a 2D tile (16 rows x 1024 B, row stride 32 KB) from the swizzled
// weight buffer into LDS. D# per CDNA5 ISA.
// 6-arg builtin form: (g0, g1, g2, g3, g4, cpol).
__device__ __forceinline__ void tdm_load_tile(unsigned lds_off, const unsigned short* gsrc) {
  unsigned long long ga = (unsigned long long)gsrc;
  u32x4 g0;
  g0.x = 1u;                                               // count=1, user mode
  g0.y = lds_off;                                          // LDS byte address
  g0.z = (unsigned)ga;                                     // global_addr[31:0]
  g0.w = (unsigned)((ga >> 32) & 0x1FFFFFFull) | (2u << 30); // addr[56:32] | type=2
  i32x8 g1;
  g1[0] = (int)(1u << 16);        // data_size = 2 bytes
  g1[1] = (int)(16384u << 16);    // tensor_dim0[15:0]=16384 in [31:16]
  g1[2] = (int)(16u << 16);       // tensor_dim1=16 in [31:16] | tensor_dim0 hi = 0
  g1[3] = (int)(512u << 16);      // tile_dim0=512 in [31:16] | tensor_dim1 hi = 0
  g1[4] = 16;                     // tile_dim2=0 | tile_dim1=16
  g1[5] = 16384;                  // tensor_dim0_stride[31:0] (elements)
  g1[6] = 0;                      // dim1_stride lo | dim0_stride hi
  g1[7] = 0;
  i32x4 z4 = {0, 0, 0, 0};
  i32x8 z8 = {0, 0, 0, 0, 0, 0, 0, 0};
  __builtin_amdgcn_tensor_load_to_lds(g0, g1, z4, z4, z8, 0);
}
#endif

// ---------------- Sinkhorn (factorized) ----------------
__global__ void k_sinkhorn(const float* __restrict__ perm, float* __restrict__ M0,
                           float* __restrict__ mM, float* __restrict__ Pm) {
  __shared__ float sred[512], sc[512], sr[512];
  const int tid = threadIdx.x;
  float mx = -3.4e38f;
  for (int i = tid; i < Hh * Hh; i += 512) mx = fmaxf(mx, perm[i]);
  sred[tid] = mx; __syncthreads();
  for (int s = 256; s > 0; s >>= 1) {
    if (tid < s) sred[tid] = fmaxf(sred[tid], sred[tid + s]);
    __syncthreads();
  }
  mx = sred[0]; __syncthreads();
  for (int i = tid; i < Hh * Hh; i += 512) M0[i] = __expf(100.0f * (perm[i] - mx));
  sc[tid] = 1.0f;
  __syncthreads();
  for (int it = 0; it < 20; ++it) {
    const float* row = M0 + (size_t)tid * Hh;
    float u = 0.0f;
    for (int j = 0; j < Hh; ++j) u += row[j] * sc[j];
    sr[tid] = 1.0f / u; __syncthreads();
    float v = 0.0f;
    for (int i = 0; i < Hh; ++i) v += M0[(size_t)i * Hh + tid] * sr[i];
    sc[tid] = 1.0f / v; __syncthreads();
  }
  const float ri = sr[tid];
  float suf = 0.0f;
  for (int j = Hh - 1; j >= 0; --j) {
    float mv = ri * M0[(size_t)tid * Hh + j] * sc[j];
    mM[(size_t)tid * Hh + j] = mv;
    suf += mv;
    Pm[(size_t)tid * Hh + j] = suf;
  }
}

// lower[i][j] = dot(m_row_i, P_row_j)    (= ((m L) m^T)^T)
__global__ void k_lower(const float* __restrict__ mM, const float* __restrict__ Pm,
                        float* __restrict__ lower) {
  __shared__ float sm[16][17], sp[16][17];
  const int tx = threadIdx.x, ty = threadIdx.y;
  const int i = blockIdx.y * 16 + ty, j = blockIdx.x * 16 + tx;
  float acc = 0.0f;
  for (int k0 = 0; k0 < Hh; k0 += 16) {
    sm[ty][tx] = mM[(size_t)i * Hh + k0 + tx];
    sp[ty][tx] = Pm[(size_t)(blockIdx.x * 16 + ty) * Hh + k0 + tx];
    __syncthreads();
#pragma unroll
    for (int kk = 0; kk < 16; ++kk) acc += sm[ty][kk] * sp[tx][kk];
    __syncthreads();
  }
  lower[(size_t)i * Hh + j] = acc;
}

// mask weights, convert to bf16, store in WMMA B-fragment order.
__global__ void k_maskswz(const float* __restrict__ w0, const float* __restrict__ w1,
                          const float* __restrict__ w2, const float* __restrict__ w3,
                          const float* __restrict__ w4, const float* __restrict__ w5,
                          const float* __restrict__ lower, unsigned short* __restrict__ wsw) {
  const float* W[6] = {w0, w1, w2, w3, w4, w5};
  int t = blockIdx.x * blockDim.x + threadIdx.x;     // 6*16*32*32 threads
  if (t >= 6 * KT * NT * 32) return;
  const int lane = t & 31, nt = (t >> 5) & 31, kt = (t >> 10) & 15, w = t >> 14;
  const int n  = nt * 16 + (lane & 15);
  const int kb = (lane & 16) ? 8 : 0;
  unsigned short out[16];
#pragma unroll
  for (int e = 0; e < 16; ++e) {
    int K = (e < 8 ? e : e + 8) + kb;                // B-fragment K mapping
    int k = kt * 32 + K;
    out[e] = f2bf(W[w][(size_t)k * Hh + n] * lower[(size_t)k * Hh + n]);
  }
  unsigned short* dst = wsw + ((((size_t)w * KT + kt) * NT + nt) * 32 + lane) * 16;
  *(v16us*)dst = *(const v16us*)out;
}

// convert X (f32) -> bf16 row-major
__global__ void k_cvtx(const float* __restrict__ x, unsigned short* __restrict__ xb) {
  size_t i0 = ((size_t)blockIdx.x * blockDim.x + threadIdx.x) * 8;
  if (i0 >= (size_t)NROW * Hh) return;
  unsigned short o[8];
#pragma unroll
  for (int e = 0; e < 8; ++e) o[e] = f2bf(x[i0 + e]);
  *(v8us*)(xb + i0) = *(const v8us*)o;
}

// Xr/Xz/Xn = X @ {Wir,Wiz,Win}_masked   (one wave per (mt,nt) tile)
__global__ void k_xproj(const unsigned short* __restrict__ xb, const unsigned short* __restrict__ wsw,
                        float* __restrict__ Xr, float* __restrict__ Xz, float* __restrict__ Xn) {
  const int wv = blockIdx.x * 8 + (threadIdx.x >> 5);   // 131072 waves
  const int mt = wv >> 5, nt = wv & 31;
  v8f ar = {0,0,0,0,0,0,0,0}, az = ar, an = ar;
#pragma unroll 4
  for (int kt = 0; kt < KT; ++kt) {
    v16bf a = load_a_frag(xb, mt * 16, kt * 32);
    ar = wmma_bf16(a, load_b_frag(wsw, 0, kt, nt), ar);
    az = wmma_bf16(a, load_b_frag(wsw, 1, kt, nt), az);
    an = wmma_bf16(a, load_b_frag(wsw, 2, kt, nt), an);
  }
  const int lane = threadIdx.x & 31;
  const int n = nt * 16 + (lane & 15), mb = (lane & 16) ? 8 : 0;
#pragma unroll
  for (int v = 0; v < 8; ++v) {
    size_t idx = (size_t)(mt * 16 + v + mb) * Hh + n;
    Xr[idx] = ar[v]; Xz[idx] = az[v]; Xn[idx] = an[v];
  }
}

__global__ void k_init(float* hcur, unsigned short* hbf, unsigned* bar) {
  int i = blockIdx.x * blockDim.x + threadIdx.x;
  if (i < Bb * Hh) { hcur[i] = 0.0f; hbf[i] = 0; }
  if (i == 0) bar[0] = 0u;
}

__device__ __forceinline__ void gbar(unsigned* bar, unsigned target) {
  __syncthreads();
  if (threadIdx.x == 0) {
    __threadfence();
    __hip_atomic_fetch_add(bar, 1u, __ATOMIC_RELEASE, __HIP_MEMORY_SCOPE_AGENT);
    while (__hip_atomic_load(bar, __ATOMIC_ACQUIRE, __HIP_MEMORY_SCOPE_AGENT) < target)
      __builtin_amdgcn_s_sleep(2);
  }
  __syncthreads();
}

// Persistent GRU scan: 64 WGs x 8 waves.
// Block b: nt = b&31, half = b>>5.  LDS holds (via TDM) two weight slices for
// this nt: slot0 = Whr (half 0) or Whz (half 1), slot1 = Whn.
// Phase 1: 8 waves -> gate tile (mt = wave, nt) for this half's gate.
// Phase 2: waves 0-3 -> n tile (mt = half*4 + wave, nt) + state update.
__global__ void k_scan(const float* __restrict__ Xr, const float* __restrict__ Xz,
                       const float* __restrict__ Xn, const unsigned short* __restrict__ wsw,
                       float* __restrict__ hcur, unsigned short* __restrict__ hbf,
                       unsigned short* __restrict__ arh, float* __restrict__ zbuf,
                       float* __restrict__ out, unsigned* __restrict__ bar) {
  __shared__ unsigned short ldsW[2 * KT * 32 * 16];       // 32 KB
  const int nt   = blockIdx.x & 31;
  const int half = blockIdx.x >> 5;                       // 0: r-gate, 1: z-gate
  const int wvL  = threadIdx.x >> 5;                      // wave in block
  const int lane = threadIdx.x & 31;
  const int mb   = (lane & 16) ? 8 : 0;

  // stage the two invariant weight slices for this block into LDS
  const size_t src0 = (((size_t)(3 + half) * KT + 0) * NT + nt) * 32 * 16;  // gate W
  const size_t src1 = (((size_t)5 * KT + 0) * NT + nt) * 32 * 16;           // Whn
#if defined(HAVE_TDM)
  if (threadIdx.x < 32) {
    unsigned lds0 = (unsigned)(unsigned long long)(const void*)ldsW;        // LDS offset
    tdm_load_tile(lds0, wsw + src0);
    tdm_load_tile(lds0 + (unsigned)(KT * 32 * 16 * 2), wsw + src1);
    __builtin_amdgcn_s_wait_tensorcnt(0);
  }
#else
  for (int i = threadIdx.x; i < KT * 32 * 16; i += 256) {
    int kt = i >> 9, j = i & 511;
    ldsW[i]                 = wsw[src0 + (size_t)kt * (NT * 32 * 16) + j];
    ldsW[KT * 32 * 16 + i]  = wsw[src1 + (size_t)kt * (NT * 32 * 16) + j];
  }
#endif
  __syncthreads();

  const float* xg = half ? Xz : Xr;
  unsigned target = 0;
  for (int t = 0; t < Tt; ++t) {
    { // ---- phase 1: this half's gate, tile (mt = wvL, nt) ----
      const int mt = wvL;
      v8f acc = {0,0,0,0,0,0,0,0};
#pragma unroll 4
      for (int kt = 0; kt < KT; ++kt)
        acc = wmma_bf16(load_a_frag(hbf, mt * 16, kt * 32),
                        load_b_lds(ldsW, 0, kt), acc);
      const int n = nt * 16 + (lane & 15);
#pragma unroll
      for (int v = 0; v < 8; ++v) {
        const int row = mt * 16 + v + mb;
        const size_t idx = (size_t)row * Hh + n;
        float pre = acc[v] + xg[(size_t)(t * Bb + row) * Hh + n];
        float g = 1.0f / (1.0f + __expf(-pre));
        if (half == 0) arh[idx] = f2bf(g * hcur[idx]);
        else           zbuf[idx] = g;
      }
    }
    target += SCAN_BLOCKS; gbar(bar, target);
    if (wvL < 4) { // ---- phase 2: n + update, tile (mt = half*4 + wvL, nt) ----
      const int mt = half * 4 + wvL;
      v8f acc = {0,0,0,0,0,0,0,0};
#pragma unroll 4
      for (int kt = 0; kt < KT; ++kt)
        acc = wmma_bf16(load_a_frag(arh, mt * 16, kt * 32),
                        load_b_lds(ldsW, 1, kt), acc);
      const int n = nt * 16 + (lane & 15);
#pragma unroll
      for (int v = 0; v < 8; ++v) {
        const int row = mt * 16 + v + mb;
        const size_t idx = (size_t)row * Hh + n;
        float pre = acc[v] + Xn[(size_t)(t * Bb + row) * Hh + n];
        float nn = tanhf(pre);
        float h = hcur[idx], z = zbuf[idx];
        float hn = h * z + (1.0f - z) * nn;
        hcur[idx] = hn; hbf[idx] = f2bf(hn);
        out[(size_t)t * (Bb * Hh) + idx] = hn;
      }
    }
    target += SCAN_BLOCKS; gbar(bar, target);
  }
}

__global__ void k_last(const int* __restrict__ lengths, float* __restrict__ out) {
  int i = blockIdx.x * blockDim.x + threadIdx.x;
  if (i >= Bb * Hh) return;
  int b = i >> 9;
  int t = lengths[b] - 1;
  out[(size_t)Tt * Bb * Hh + i] = out[(size_t)t * (Bb * Hh) + i];
}

extern "C" void kernel_launch(void* const* d_in, const int* in_sizes, int n_in,
                              void* d_out, int out_size, void* d_ws, size_t ws_size,
                              hipStream_t stream) {
  const float* input_  = (const float*)d_in[0];
  const int*   lengths = (const int*)d_in[1];
  const float* perm    = (const float*)d_in[2];
  const float* W_ir = (const float*)d_in[3];
  const float* W_hr = (const float*)d_in[4];
  const float* W_iz = (const float*)d_in[5];
  const float* W_hz = (const float*)d_in[6];
  const float* W_in = (const float*)d_in[7];
  const float* W_hn = (const float*)d_in[8];

  char* ws = (char*)d_ws;
  float*          M0    = (float*)(ws + OFF_M0);
  float*          mM    = (float*)(ws + OFF_M);
  float*          Pm    = (float*)(ws + OFF_P);
  float*          lower = (float*)(ws + OFF_LOWER);
  unsigned short* wsw   = (unsigned short*)(ws + OFF_WSW);
  unsigned short* xb    = (unsigned short*)(ws + OFF_XB);
  float*          Xr    = (float*)(ws + OFF_XR);
  float*          Xz    = (float*)(ws + OFF_XZ);
  float*          Xn    = (float*)(ws + OFF_XN);
  float*          hcur  = (float*)(ws + OFF_H);
  unsigned short* hbf   = (unsigned short*)(ws + OFF_HBF);
  unsigned short* arh   = (unsigned short*)(ws + OFF_ARH);
  float*          zbuf  = (float*)(ws + OFF_Z);
  unsigned*       bar   = (unsigned*)(ws + OFF_BAR);
  float*          out   = (float*)d_out;

  k_sinkhorn<<<1, 512, 0, stream>>>(perm, M0, mM, Pm);
  k_lower<<<dim3(32, 32), dim3(16, 16), 0, stream>>>(mM, Pm, lower);
  k_maskswz<<<(6 * KT * NT * 32 + 255) / 256, 256, 0, stream>>>(
      W_ir, W_iz, W_in, W_hr, W_hz, W_hn, lower, wsw);
  k_cvtx<<<(int)(((size_t)NROW * Hh / 8 + 255) / 256), 256, 0, stream>>>(input_, xb);
  k_xproj<<<(NROW / 16) * NT / 8, 256, 0, stream>>>(xb, wsw, Xr, Xz, Xn);
  k_init<<<(Bb * Hh + 255) / 256, 256, 0, stream>>>(hcur, hbf, bar);
  k_scan<<<SCAN_BLOCKS, 256, 0, stream>>>(Xr, Xz, Xn, wsw, hcur, hbf, arh, zbuf, out, bar);
  k_last<<<(Bb * Hh + 255) / 256, 256, 0, stream>>>(lengths, out);
}
